// InteractionModule_21191368639081
// MI455X (gfx1250) — compile-verified, matrix operand order checked by reading, not run on hardware
//
#include <hip/hip_runtime.h>
#include <hip/hip_bf16.h>

// -------- problem constants (from the reference) --------
#define F       128
#define KDIM    64
#define RLAYERS 3
#define NNODES  50000
#define NEDGES  600000

typedef __attribute__((ext_vector_type(16))) __bf16 v16bf;
typedef __attribute__((ext_vector_type(8)))  float  v8f;

__device__ __forceinline__ float ssp_f(float x) {
  // shifted softplus: log(1+exp(x)) - log(2), numerically stable
  float ax = fabsf(x);
  return fmaxf(x, 0.0f) + log1pf(__expf(-ax)) - 0.69314718055994530942f;
}

__device__ __forceinline__ v8f zero8() {
  v8f z;
#pragma unroll
  for (int j = 0; j < 8; ++j) z[j] = 0.0f;
  return z;
}

// Load one 16x32 bf16 operand fragment from a row-major bf16 matrix.
// Element (row0+m, k0+k) using the CDNA5 §7.12.2 per-lane swizzle:
//   lanes 0..15  own rows m=lane   with K in {0..7, 16..23}
//   lanes 16..31 own rows m=lane-16 with K in {8..15, 24..31}
// For the B operand we pass W row-major [F_out][F_in]; since B[k][n] = W[n][k],
// "row" = output column n and the per-lane K pairs are contiguous dwords.
__device__ __forceinline__ v16bf load_frag(const __bf16* base, int row0, int ld,
                                           int k0, int lane) {
  int m  = lane & 15;
  int kh = (lane >> 4) & 1;
  const __bf16* rp = base + (size_t)(row0 + m) * ld + k0 + 8 * kh;
  v16bf v;
#pragma unroll
  for (int j = 0; j < 16; j += 2) {
    int k = (j & 7) + ((j >> 3) << 4);          // 0..7 then 16..23 (lane-half adds 8)
    unsigned int d = *(const unsigned int*)(rp + k);
    v[j]     = ((const __bf16*)&d)[0];
    v[j + 1] = ((const __bf16*)&d)[1];
  }
  return v;
}

// -------- prep: xa = ssp(x) in bf16; zero the aggr/msged accumulator --------
__global__ __launch_bounds__(256) void prep_nodes(const float* __restrict__ x,
                                                  __bf16* __restrict__ xa_bf,
                                                  float* __restrict__ aggr,
                                                  size_t n) {
  size_t i = (size_t)blockIdx.x * 256 + threadIdx.x;
  if (i < n) {
    xa_bf[i] = (__bf16)ssp_f(x[i]);
    aggr[i]  = 0.0f;   // aggr lives in d_out[NF..2NF): must be re-zeroed every call
  }
}

// -------- prep: convert all weights to bf16, kept row-major [F_out][F_in] ----
// wbf layout: [0]=W_same FF | [FF]=W_diff | [2FF]=W_last | [3FF]=res_W1 (3*FF)
//             [6FF]=res_W2 (3*FF) | [9FF]=G_w (F*K)
__global__ __launch_bounds__(256) void prep_weights(
    const float* __restrict__ Wsame, const float* __restrict__ Wdiff,
    const float* __restrict__ Wlast, const float* __restrict__ rW1,
    const float* __restrict__ rW2,   const float* __restrict__ Gw,
    __bf16* __restrict__ wbf) {
  int i = blockIdx.x * 256 + threadIdx.x;
  const int FF = F * F;
  if      (i <     FF)            wbf[i] = (__bf16)Wsame[i];
  else if (i < 2 * FF)            wbf[i] = (__bf16)Wdiff[i - FF];
  else if (i < 3 * FF)            wbf[i] = (__bf16)Wlast[i - 2 * FF];
  else if (i < 6 * FF)            wbf[i] = (__bf16)rW1[i - 3 * FF];
  else if (i < 9 * FF)            wbf[i] = (__bf16)rW2[i - 6 * FF];
  else if (i < 9 * FF + F * KDIM) wbf[i] = (__bf16)Gw[i - 9 * FF];
}

// -------- edge kernel: msg = ssp(xa[src]@Wd^T + bd) * (ea@Gw^T); scatter-add --
__global__ __launch_bounds__(256) void edge_kernel(
    const __bf16* __restrict__ xa_bf, const int* __restrict__ edge_index,
    const float* __restrict__ edge_attr, const __bf16* __restrict__ wbf,
    const float* __restrict__ b_diff, float* __restrict__ aggr) {
  __shared__ __bf16 sX[64][132];   // gathered xa rows (pad 132 -> conflict-free frags)
  __shared__ __bf16 sEA[64][68];   // edge_attr tile in bf16 (pad 68)
  __shared__ int    sDst[64];

  const int tid = threadIdx.x, lane = tid & 31, wave = tid >> 5;
  const int mb = wave >> 1;            // M band 0..3 (16 edges each)
  const int ntb = (wave & 1) * 4;      // 4 N tiles per wave
  const int eBase = blockIdx.x * 64;   // E = 9375 * 64, exact

  const __bf16* Wdiff = wbf + (size_t)F * F;
  const __bf16* Gw    = wbf + 9 * (size_t)F * F;

  { // stage: 4 threads per edge row
    int el = tid >> 2, part = tid & 3;
    int e = eBase + el;
    int src = edge_index[e];
    const unsigned int* gp = (const unsigned int*)(xa_bf + (size_t)src * F + part * 32);
    unsigned int* lp = (unsigned int*)&sX[el][part * 32];
#pragma unroll
    for (int q = 0; q < 16; ++q) lp[q] = gp[q];
    const float* ep = edge_attr + (size_t)e * KDIM + part * 16;
#pragma unroll
    for (int q = 0; q < 16; ++q) sEA[el][part * 16 + q] = (__bf16)ep[q];
    if (part == 0) sDst[el] = edge_index[NEDGES + e];
  }
  __syncthreads();

  v8f gacc[4], macc[4];
#pragma unroll
  for (int i = 0; i < 4; ++i) { gacc[i] = zero8(); macc[i] = zero8(); }

  // gate = ea @ G_w^T  (K = 64)
  for (int k0 = 0; k0 < KDIM; k0 += 32) {
    v16bf a = load_frag(&sEA[0][0], mb * 16, 68, k0, lane);
#pragma unroll
    for (int i = 0; i < 4; ++i) {
      v16bf b = load_frag(Gw, (ntb + i) * 16, KDIM, k0, lane);
      gacc[i] = __builtin_amdgcn_wmma_f32_16x16x32_bf16(false, a, false, b,
                                                        (short)0, gacc[i], false, false);
    }
  }
  // z = xa_gathered @ W_diff^T  (K = 128)
  for (int k0 = 0; k0 < F; k0 += 32) {
    v16bf a = load_frag(&sX[0][0], mb * 16, 132, k0, lane);
#pragma unroll
    for (int i = 0; i < 4; ++i) {
      v16bf b = load_frag(Wdiff, (ntb + i) * 16, F, k0, lane);
      macc[i] = __builtin_amdgcn_wmma_f32_16x16x32_bf16(false, a, false, b,
                                                        (short)0, macc[i], false, false);
    }
  }
  // msg = ssp(z + b_diff) * gate ; scatter-add into aggr[dst]
#pragma unroll
  for (int i = 0; i < 4; ++i) {
    int col = (ntb + i) * 16 + (lane & 15);
    float bd = b_diff[col];
#pragma unroll
    for (int r = 0; r < 8; ++r) {
      int rl = mb * 16 + r + 8 * (lane >> 4);
      float msg = ssp_f(macc[i][r] + bd) * gacc[i][r];
      atomicAdd(&aggr[(size_t)sDst[rl] * F + col], msg);
    }
  }
}

// -------- node kernel: self GEMM + aggr, residual stack, last layer, +x*u ----
__global__ __launch_bounds__(256) void node_kernel(
    const __bf16* __restrict__ xa_bf, const __bf16* __restrict__ wbf,
    const float* __restrict__ b_same, const float* __restrict__ res_b1,
    const float* __restrict__ res_b2, const float* __restrict__ b_last,
    const float* __restrict__ x, const float* __restrict__ u,
    float* __restrict__ out) {           // out[0..NF)=v+x*u ; out[NF..2NF)=msged_x
  __shared__ __bf16 sA[64][132];         // bf16 activation staging
  __shared__ float  tF[64][F];           // running t (fp32)

  const int tid = threadIdx.x, lane = tid & 31, wave = tid >> 5;
  const int mb = wave >> 1, ntb = (wave & 1) * 4;
  const int rowBase = blockIdx.x * 64;
  const size_t NF = (size_t)NNODES * F;
  float* msged = out + NF;               // also holds the edge aggregation

  const __bf16* Wsame = wbf;
  const __bf16* Wlast = wbf + 2 * (size_t)F * F;
  const __bf16* rW1   = wbf + 3 * (size_t)F * F;
  const __bf16* rW2   = wbf + 6 * (size_t)F * F;

  { // stage xa tile (zero-fill past N)
    int r = tid >> 2, cb = (tid & 3) * 32;
    int gr = rowBase + r;
    unsigned int* lp = (unsigned int*)&sA[r][cb];
    if (gr < NNODES) {
      const unsigned int* gp = (const unsigned int*)(xa_bf + (size_t)gr * F + cb);
#pragma unroll
      for (int q = 0; q < 16; ++q) lp[q] = gp[q];
    } else {
#pragma unroll
      for (int q = 0; q < 16; ++q) lp[q] = 0u;
    }
  }
  __syncthreads();

  v8f acc[4];
  // ---- t0 = ssp(xa @ W_same^T + b_same) + aggr ; also emit msged_x output
#pragma unroll
  for (int i = 0; i < 4; ++i) acc[i] = zero8();
  for (int k0 = 0; k0 < F; k0 += 32) {
    v16bf a = load_frag(&sA[0][0], mb * 16, 132, k0, lane);
#pragma unroll
    for (int i = 0; i < 4; ++i) {
      v16bf b = load_frag(Wsame, (ntb + i) * 16, F, k0, lane);
      acc[i] = __builtin_amdgcn_wmma_f32_16x16x32_bf16(false, a, false, b,
                                                       (short)0, acc[i], false, false);
    }
  }
#pragma unroll
  for (int i = 0; i < 4; ++i) {
    int col = (ntb + i) * 16 + (lane & 15);
    float bs = b_same[col];
#pragma unroll
    for (int r = 0; r < 8; ++r) {
      int rl = mb * 16 + r + 8 * (lane >> 4);
      int gr = rowBase + rl;
      float ag = (gr < NNODES) ? msged[(size_t)gr * F + col] : 0.0f;  // aggr
      float t  = ssp_f(acc[i][r] + bs) + ag;
      tF[rl][col] = t;
      if (gr < NNODES) msged[(size_t)gr * F + col] = t;               // msged_x out
    }
  }
  __syncthreads();

  // ---- residual stack: t += ssp( ssp(t)@W1^T + b1 ) @ W2^T + b2
  for (int layer = 0; layer < RLAYERS; ++layer) {
    { // sA = bf16(ssp(t))
      int r = tid >> 2, cb = (tid & 3) * 32;
#pragma unroll
      for (int q = 0; q < 32; ++q) sA[r][cb + q] = (__bf16)ssp_f(tF[r][cb + q]);
    }
    __syncthreads();
#pragma unroll
    for (int i = 0; i < 4; ++i) acc[i] = zero8();
    const __bf16* W1 = rW1 + (size_t)layer * F * F;
    for (int k0 = 0; k0 < F; k0 += 32) {
      v16bf a = load_frag(&sA[0][0], mb * 16, 132, k0, lane);
#pragma unroll
      for (int i = 0; i < 4; ++i) {
        v16bf b = load_frag(W1, (ntb + i) * 16, F, k0, lane);
        acc[i] = __builtin_amdgcn_wmma_f32_16x16x32_bf16(false, a, false, b,
                                                         (short)0, acc[i], false, false);
      }
    }
    __syncthreads();   // all sA reads done; safe to overwrite with ssp(y1)
#pragma unroll
    for (int i = 0; i < 4; ++i) {
      int col = (ntb + i) * 16 + (lane & 15);
      float b1 = res_b1[layer * F + col];
#pragma unroll
      for (int r = 0; r < 8; ++r) {
        int rl = mb * 16 + r + 8 * (lane >> 4);
        sA[rl][col] = (__bf16)ssp_f(acc[i][r] + b1);
      }
    }
    __syncthreads();
#pragma unroll
    for (int i = 0; i < 4; ++i) acc[i] = zero8();
    const __bf16* W2 = rW2 + (size_t)layer * F * F;
    for (int k0 = 0; k0 < F; k0 += 32) {
      v16bf a = load_frag(&sA[0][0], mb * 16, 132, k0, lane);
#pragma unroll
      for (int i = 0; i < 4; ++i) {
        v16bf b = load_frag(W2, (ntb + i) * 16, F, k0, lane);
        acc[i] = __builtin_amdgcn_wmma_f32_16x16x32_bf16(false, a, false, b,
                                                         (short)0, acc[i], false, false);
      }
    }
#pragma unroll
    for (int i = 0; i < 4; ++i) {
      int col = (ntb + i) * 16 + (lane & 15);
      float b2 = res_b2[layer * F + col];
#pragma unroll
      for (int r = 0; r < 8; ++r) {
        int rl = mb * 16 + r + 8 * (lane >> 4);
        tF[rl][col] += acc[i][r] + b2;    // each element owned by exactly one lane
      }
    }
    __syncthreads();
  }

  // ---- v = ssp(t) @ W_last^T + b_last ; out = v + x*u
  {
    int r = tid >> 2, cb = (tid & 3) * 32;
#pragma unroll
    for (int q = 0; q < 32; ++q) sA[r][cb + q] = (__bf16)ssp_f(tF[r][cb + q]);
  }
  __syncthreads();
#pragma unroll
  for (int i = 0; i < 4; ++i) acc[i] = zero8();
  for (int k0 = 0; k0 < F; k0 += 32) {
    v16bf a = load_frag(&sA[0][0], mb * 16, 132, k0, lane);
#pragma unroll
    for (int i = 0; i < 4; ++i) {
      v16bf b = load_frag(Wlast, (ntb + i) * 16, F, k0, lane);
      acc[i] = __builtin_amdgcn_wmma_f32_16x16x32_bf16(false, a, false, b,
                                                       (short)0, acc[i], false, false);
    }
  }
#pragma unroll
  for (int i = 0; i < 4; ++i) {
    int col = (ntb + i) * 16 + (lane & 15);
    float bl = b_last[col];
    float uc = u[col];
#pragma unroll
    for (int r = 0; r < 8; ++r) {
      int rl = mb * 16 + r + 8 * (lane >> 4);
      int gr = rowBase + rl;
      if (gr < NNODES)
        out[(size_t)gr * F + col] = acc[i][r] + bl + x[(size_t)gr * F + col] * uc;
    }
  }
}

extern "C" void kernel_launch(void* const* d_in, const int* in_sizes, int n_in,
                              void* d_out, int out_size, void* d_ws, size_t ws_size,
                              hipStream_t stream) {
  (void)in_sizes; (void)n_in; (void)out_size; (void)ws_size;
  const float* x     = (const float*)d_in[0];
  const int*   ei    = (const int*)d_in[1];
  const float* ea    = (const float*)d_in[2];
  const float* Wsame = (const float*)d_in[3];
  const float* bsame = (const float*)d_in[4];
  const float* Wdiff = (const float*)d_in[5];
  const float* bdiff = (const float*)d_in[6];
  const float* Gw    = (const float*)d_in[7];
  const float* rW1   = (const float*)d_in[8];
  const float* rb1   = (const float*)d_in[9];
  const float* rW2   = (const float*)d_in[10];
  const float* rb2   = (const float*)d_in[11];
  const float* Wlast = (const float*)d_in[12];
  const float* blast = (const float*)d_in[13];
  const float* u     = (const float*)d_in[14];
  float* out = (float*)d_out;

  const size_t NF = (size_t)NNODES * F;
  // workspace: xa_bf (N*F bf16) | weights_bf (9*F*F + F*K bf16)  ~13.1 MB
  __bf16* xa_bf = (__bf16*)d_ws;
  __bf16* wbf   = xa_bf + NF;
  float*  aggr  = out + NF;   // aggr accumulates in the msged_x output slot

  prep_nodes<<<(unsigned)((NF + 255) / 256), 256, 0, stream>>>(x, xa_bf, aggr, NF);
  const int wtot = 9 * F * F + F * KDIM;
  prep_weights<<<(wtot + 255) / 256, 256, 0, stream>>>(Wsame, Wdiff, Wlast, rW1, rW2, Gw, wbf);
  edge_kernel<<<NEDGES / 64, 256, 0, stream>>>(xa_bf, ei, ea, wbf, bdiff, aggr);
  node_kernel<<<(NNODES + 63) / 64, 256, 0, stream>>>(xa_bf, wbf, bsame, rb1, rb2,
                                                      blast, x, u, out);
}
